// FastARGO_38070590112284
// MI455X (gfx1250) — compile-verified, matrix operand order checked by reading, not run on hardware
//
#include <hip/hip_runtime.h>
#include <hip/hip_bf16.h>
#include <math.h>

// ---------------- types / WMMA helpers ----------------
typedef __attribute__((ext_vector_type(16))) _Float16 v16h;
typedef __attribute__((ext_vector_type(8)))  _Float16 v8h;
typedef __attribute__((ext_vector_type(8)))  float    v8f;

union FragU { v16h v; v8h h[2]; };

#define WMMA_F16(A, Bf, C) \
  __builtin_amdgcn_wmma_f32_16x16x32_f16(false, (A), false, (Bf), (short)0, (C), false, false)

#if __has_builtin(__builtin_amdgcn_rcpf)
#define FAST_RCP(x) __builtin_amdgcn_rcpf(x)
#else
#define FAST_RCP(x) (1.f / (x))
#endif

#if __has_builtin(__builtin_amdgcn_tanhf)
#define FAST_TANH(x) __builtin_amdgcn_tanhf(x)
#elif __has_builtin(__builtin_amdgcn_tanh_f32)
#define FAST_TANH(x) __builtin_amdgcn_tanh_f32(x)
#else
#define FAST_TANH(x) tanhf(x)
#endif

__device__ __forceinline__ float fast_sigmoid(float x) {
  return FAST_RCP(1.f + __expf(-x));
}

// Problem constants: B=8, N=512, T=64, D=128, E=32, M=3, BN=4096, TOPK=25

// ---------------- setup kernels ----------------
__global__ void uv_kernel(const float* __restrict__ Wih, const float* __restrict__ pW,
                          const float* __restrict__ pb, const float* __restrict__ bih,
                          float* __restrict__ u, float* __restrict__ v) {
  int j = blockIdx.x * 128 + threadIdx.x;
  if (j >= 384) return;
  float su = 0.f, sv = 0.f;
  for (int d = 0; d < 128; ++d) {
    float w = Wih[j * 128 + d];
    su += w * pW[d];
    sv += w * pb[d];
  }
  u[j] = su;
  v[j] = sv + bih[j];
}

__global__ void cvt_f16_kernel(const float* __restrict__ src, _Float16* __restrict__ dst, int n) {
  int i = blockIdx.x * 256 + threadIdx.x;
  if (i < n) dst[i] = (_Float16)src[i];
}

__global__ __launch_bounds__(256) void qnorm_kernel(const float* __restrict__ q,
                                                    float* __restrict__ qn) {
  int lane = threadIdx.x & 31, wid = threadIdx.x >> 5;
  int row = blockIdx.x * 8 + wid;           // 0..95  (= M*E)
  const float* r = q + row * 128;
  float a[4]; float ss = 0.f;
#pragma unroll
  for (int i = 0; i < 4; ++i) { a[i] = r[lane + 32 * i]; ss += a[i] * a[i]; }
#pragma unroll
  for (int o = 16; o > 0; o >>= 1) ss += __shfl_xor(ss, o, 32);
  float scale = 1.f / fmaxf(sqrtf(ss), 1e-12f);
#pragma unroll
  for (int i = 0; i < 4; ++i) qn[row * 128 + lane + 32 * i] = a[i] * scale;
}

// ---------------- GRU (WMMA) ----------------
// grid 32 x 256 threads (8 waves); wave owns 16 sequences for all 64 steps.
__global__ __launch_bounds__(256) void gru_kernel(
    const float* __restrict__ x, const float* __restrict__ uvec,
    const float* __restrict__ vvec, const float* __restrict__ bhh,
    const _Float16* __restrict__ whh16,
    float* __restrict__ hout, _Float16* __restrict__ hT) {
  __shared__ _Float16 sW[384 * 128];     // W_hh row-major f16 (96KB)
  __shared__ _Float16 sH[8][16 * 128];   // per-wave h tile, row-major [s][d] f16
  __shared__ float sU[384], sV[384], sB[384];  // gate coefficients

  const int tid = threadIdx.x, lane = tid & 31, wid = tid >> 5;
  const int col = lane & 15, hi = lane >> 4;

  {  // cooperative W_hh load
    const v8h* src = (const v8h*)whh16;
    v8h* dst = (v8h*)sW;
    for (int i = tid; i < 384 * 128 / 8; i += 256) dst[i] = src[i];
  }
  for (int i = tid; i < 384; i += 256) {
    sU[i] = uvec[i];
    sV[i] = vvec[i];
    sB[i] = bhh[i];
  }
  _Float16* hRow = &sH[wid][0];
  {  // zero this wave's h rows
    v8h z = {};
    for (int i = lane; i < 16 * 128 / 8; i += 32) ((v8h*)hRow)[i] = z;
  }
  __syncthreads();

  v8f hC[8];
#pragma unroll
  for (int db = 0; db < 8; ++db) hC[db] = {};

  const int wavebase = blockIdx.x * 128 + wid * 16;
  const int sLane = wavebase + col;          // sequence index for this lane's x fetch
  const int xb = (sLane >> 9) << 15;         // b * N * T
  const int rr0 = (sLane & 511) << 6;

  for (int u = 0; u < 64; ++u) {
    // Prevent LICM from hoisting the (loop-invariant) LDS W/coef loads out of
    // the 64-step loop -- that caused massive scratch spills otherwise.
    asm volatile("" ::: "memory");

    const int rr = rr0 + u;
    float xv = x[xb + ((rr & 511) << 6) + (rr >> 9)];
    if (u + 1 < 64) {
      const int rn = rr + 1;
      __builtin_prefetch(&x[xb + ((rn & 511) << 6) + (rn >> 9)], 0, 1);
    }
    float xs[8];
#pragma unroll
    for (int r2 = 0; r2 < 8; ++r2) xs[r2] = __shfl(xv, (hi << 3) + r2, 32);

    // A fragments of h (16 x 32 per kblk)
    FragU a[4];
#pragma unroll
    for (int kb = 0; kb < 4; ++kb) {
      const int base = kb * 32 + hi * 8;
      a[kb].h[0] = *(const v8h*)&hRow[col * 128 + base];
      a[kb].h[1] = *(const v8h*)&hRow[col * 128 + base + 16];
    }

#pragma unroll
    for (int db = 0; db < 8; ++db) {
      v8f ar = {}, az = {}, an = {};
      const int j0 = (db * 16 + col) * 128;
#pragma unroll
      for (int kb = 0; kb < 4; ++kb) {
        const int wb = kb * 32 + hi * 8;
        FragU bf;
        bf.h[0] = *(const v8h*)&sW[j0 + wb];
        bf.h[1] = *(const v8h*)&sW[j0 + wb + 16];
        ar = WMMA_F16(a[kb].v, bf.v, ar);
        bf.h[0] = *(const v8h*)&sW[j0 + 128 * 128 + wb];
        bf.h[1] = *(const v8h*)&sW[j0 + 128 * 128 + wb + 16];
        az = WMMA_F16(a[kb].v, bf.v, az);
        bf.h[0] = *(const v8h*)&sW[j0 + 256 * 128 + wb];
        bf.h[1] = *(const v8h*)&sW[j0 + 256 * 128 + wb + 16];
        an = WMMA_F16(a[kb].v, bf.v, an);
      }
      // gate coefficients: short live range (one dblk iteration)
      const int j = db * 16 + col;
      const float urv = sU[j], uzv = sU[128 + j], unv = sU[256 + j];
      const float vrv = sV[j], vzv = sV[128 + j], vnv = sV[256 + j];
      const float brv = sB[j], bzv = sB[128 + j], bnv = sB[256 + j];
      // gate math: C-layout register r2 maps to (s = base+hi*8+r2, d = db*16+col)
#pragma unroll
      for (int r2 = 0; r2 < 8; ++r2) {
        float rg = fast_sigmoid(xs[r2] * urv + vrv + ar[r2] + brv);
        float zg = fast_sigmoid(xs[r2] * uzv + vzv + az[r2] + bzv);
        float ng = FAST_TANH(xs[r2] * unv + vnv + rg * (an[r2] + bnv));
        float hn2 = (1.f - zg) * ng + zg * hC[db][r2];
        hC[db][r2] = hn2;
        hRow[(hi * 8 + r2) * 128 + db * 16 + col] = (_Float16)hn2;
      }
    }
  }

  // final hidden: h f32 [4096][128] and hT f16 [8][128][512]
#pragma unroll
  for (int db = 0; db < 8; ++db) {
#pragma unroll
    for (int r2 = 0; r2 < 8; ++r2) {
      int s = wavebase + hi * 8 + r2;
      int d = db * 16 + col;
      hout[s * 128 + d] = hC[db][r2];
      int b = s >> 9, n = s & 511;
      hT[(((size_t)b * 128) + d) * 512 + n] = (_Float16)hC[db][r2];
    }
  }
}

// ---------------- stats + router ----------------
__global__ __launch_bounds__(256) void stats_kernel(
    const float* __restrict__ h, const float* __restrict__ rW, const float* __restrict__ rb,
    float* __restrict__ probs, float* __restrict__ outProbs) {
  __shared__ float st[256];
  __shared__ float red[3][256];
  int b = blockIdx.x, tid = threadIdx.x;
  if (tid < 128) {
    float s = 0.f, s2 = 0.f;
    for (int n = 0; n < 512; ++n) {
      float v = h[((size_t)(b * 512 + n)) * 128 + tid];
      s += v; s2 += v * v;
    }
    float mean = s / 512.f;
    float var = (s2 - s * mean) / 511.f;   // unbiased
    st[tid] = mean;
    st[128 + tid] = sqrtf(fmaxf(var, 0.f));
  }
  __syncthreads();
  float sv = st[tid];
  for (int m = 0; m < 3; ++m) red[m][tid] = sv * rW[m * 256 + tid];
  __syncthreads();
  for (int off = 128; off > 0; off >>= 1) {
    if (tid < off)
      for (int m = 0; m < 3; ++m) red[m][tid] += red[m][tid + off];
    __syncthreads();
  }
  if (tid == 0) {
    float l0 = red[0][0] + rb[0], l1 = red[1][0] + rb[1], l2 = red[2][0] + rb[2];
    float mx = fmaxf(l0, fmaxf(l1, l2));
    float e0 = __expf(l0 - mx), e1 = __expf(l1 - mx), e2 = __expf(l2 - mx);
    float s = e0 + e1 + e2;
    float p0 = e0 / s, p1 = e1 / s, p2 = e2 / s;
    probs[b * 3 + 0] = p0; probs[b * 3 + 1] = p1; probs[b * 3 + 2] = p2;
    outProbs[b * 3 + 0] = p0; outProbs[b * 3 + 1] = p1; outProbs[b * 3 + 2] = p2;
  }
}

// ---------------- z_prior = prior @ h  (WMMA) ----------------
__global__ __launch_bounds__(128) void zprior_kernel(
    const _Float16* __restrict__ pr16, const _Float16* __restrict__ hT,
    float* __restrict__ zp) {
  const int lane = threadIdx.x & 31, wid = threadIdx.x >> 5;
  const int w = blockIdx.x * 4 + wid;     // 0..2047
  const int nt = w & 7, mt = (w >> 3) & 31, b = w >> 8;
  const int col = lane & 15, hi = lane >> 4;
  const _Float16* hTb = hT + (size_t)b * 128 * 512;
  v8f acc = {};
  const int rowA = (mt * 16 + col) * 512;
  const int rowB = (nt * 16 + col) * 512;
#pragma unroll 4
  for (int k = 0; k < 16; ++k) {
    const int ka = k * 32 + hi * 8;
    FragU a, bf;
    a.h[0]  = *(const v8h*)&pr16[rowA + ka];
    a.h[1]  = *(const v8h*)&pr16[rowA + ka + 16];
    bf.h[0] = *(const v8h*)&hTb[rowB + ka];
    bf.h[1] = *(const v8h*)&hTb[rowB + ka + 16];
    acc = WMMA_F16(a.v, bf.v, acc);
  }
#pragma unroll
  for (int r2 = 0; r2 < 8; ++r2) {
    int row = mt * 16 + hi * 8 + r2;
    int d = nt * 16 + col;
    zp[((size_t)b * 512 + row) * 128 + d] = acc[r2];
  }
}

// ---------------- hs = norm(h@spW^T + spB); scores = hs . qn ----------------
__global__ __launch_bounds__(256) void scores_kernel(
    const float* __restrict__ h, const float* __restrict__ spW, const float* __restrict__ spB,
    const float* __restrict__ qn, float* __restrict__ scores) {
  __shared__ float hsld[8][128];
  int lane = threadIdx.x & 31, wid = threadIdx.x >> 5;
  int w = blockIdx.x * 8 + wid;          // 0..12287  (= B*M*N)
  int n = w & 511, m = (w >> 9) % 3, b = w / (512 * 3);
  const float* hrow = h + ((size_t)(b * 512 + n)) * 128;
  const float* Wm = spW + m * 128 * 128;
  float acc[4] = {0.f, 0.f, 0.f, 0.f};
  for (int d = 0; d < 128; ++d) {
    float hv = hrow[d];
#pragma unroll
    for (int i = 0; i < 4; ++i) acc[i] += hv * Wm[(lane + 32 * i) * 128 + d];
  }
  float ss = 0.f;
#pragma unroll
  for (int i = 0; i < 4; ++i) {
    acc[i] += spB[m * 128 + lane + 32 * i];
    ss += acc[i] * acc[i];
  }
#pragma unroll
  for (int o = 16; o > 0; o >>= 1) ss += __shfl_xor(ss, o, 32);
  float scale = 1.f / fmaxf(sqrtf(ss), 1e-12f);
#pragma unroll
  for (int i = 0; i < 4; ++i) hsld[wid][lane + 32 * i] = acc[i] * scale;
  __syncthreads();
  float sc = 0.f;
  const float* q = qn + (m * 32 + lane) * 128;
  for (int k = 0; k < 128; ++k) sc += hsld[wid][k] * q[k];
  scores[((size_t)((b * 3 + m) * 32 + lane)) * 512 + n] = sc;
}

// ---------------- top-25 + softmax per (b,m,e) row ----------------
__global__ __launch_bounds__(256) void topk_kernel(
    const float* __restrict__ scores, int* __restrict__ Hidx, float* __restrict__ Hw) {
  __shared__ float tv[8][25];
  __shared__ int ti[8][25];
  int lane = threadIdx.x & 31, wid = threadIdx.x >> 5;
  int row = blockIdx.x * 8 + wid;   // 0..767
  const float* sr = scores + (size_t)row * 512;
  float sv[16];
#pragma unroll
  for (int i = 0; i < 16; ++i) sv[i] = sr[lane * 16 + i];
  for (int it = 0; it < 25; ++it) {
    float mv = -1e30f; int mi = 0;
#pragma unroll
    for (int i = 0; i < 16; ++i) {
      int idx = lane * 16 + i;
      if (sv[i] > mv || (sv[i] == mv && idx < mi)) { mv = sv[i]; mi = idx; }
    }
#pragma unroll
    for (int o = 16; o > 0; o >>= 1) {
      float ov = __shfl_xor(mv, o, 32); int oi = __shfl_xor(mi, o, 32);
      if (ov > mv || (ov == mv && oi < mi)) { mv = ov; mi = oi; }
    }
    if ((mi >> 4) == lane) sv[mi & 15] = -3e38f;
    if (lane == 0) { tv[wid][it] = mv; ti[wid][it] = mi; }
  }
  __syncthreads();
  float vmax = tv[wid][0];
  float ex = (lane < 25) ? __expf((tv[wid][lane] - vmax) * (1.f / 0.7f)) : 0.f;
  float sum = ex;
#pragma unroll
  for (int o = 16; o > 0; o >>= 1) sum += __shfl_xor(sum, o, 32);
  if (lane < 25) {
    Hw[row * 25 + lane] = ex / sum;
    Hidx[row * 25 + lane] = ti[wid][lane];
  }
}

// ---------------- dense Hmix[b][e][n] = sum_m probs * H ----------------
__global__ void hmix_kernel(const int* __restrict__ Hidx, const float* __restrict__ Hw,
                            const float* __restrict__ probs, float* __restrict__ Hmix) {
  int t = blockIdx.x * 256 + threadIdx.x;   // 0..131071
  int n = t & 511, e = (t >> 9) & 31, b = t >> 14;
  float acc = 0.f;
  for (int m = 0; m < 3; ++m) {
    float pm = probs[b * 3 + m];
    int rb = ((b * 3 + m) * 32 + e) * 25;
    for (int i = 0; i < 25; ++i)
      if (Hidx[rb + i] == n) acc += pm * Hw[rb + i];
  }
  Hmix[t] = acc;
}

// ---------------- edge[b][e][d] = sum_n Hmix * h ----------------
__global__ __launch_bounds__(128) void edge_kernel(
    const float* __restrict__ Hmix, const float* __restrict__ h, float* __restrict__ edge) {
  int be = blockIdx.x;               // 0..255
  int b = be >> 5;
  int d = threadIdx.x;
  const float* Hm = Hmix + (size_t)be * 512;
  const float* hb = h + (size_t)b * 512 * 128;
  float acc = 0.f;
  for (int n = 0; n < 512; ++n) {
    float w = Hm[n];
    if (w != 0.f) acc += w * hb[n * 128 + d];
  }
  edge[(size_t)be * 128 + d] = acc;
}

// ---------------- z_learn + blend + out projection ----------------
__global__ __launch_bounds__(256) void final_kernel(
    const float* __restrict__ Hmix, const float* __restrict__ edge,
    const float* __restrict__ zp, const float* __restrict__ outW,
    const float* __restrict__ outb, const float* __restrict__ plog,
    float* __restrict__ out) {
  int lane = threadIdx.x & 31, wid = threadIdx.x >> 5;
  int w = blockIdx.x * 8 + wid;      // 0..4095
  int b = w >> 9, n = w & 511;
  float alpha = fast_sigmoid(plog[0]);
  float acc = 0.f;
#pragma unroll
  for (int i = 0; i < 4; ++i) {
    int d = lane + 32 * i;
    float zl = 0.f;
    for (int e = 0; e < 32; ++e)
      zl += Hmix[((size_t)(b * 32 + e)) * 512 + n] * edge[((size_t)(b * 32 + e)) * 128 + d];
    float mix = alpha * zp[((size_t)(b * 512 + n)) * 128 + d] + (1.f - alpha) * zl;
    acc += mix * outW[d];
  }
#pragma unroll
  for (int o = 16; o > 0; o >>= 1) acc += __shfl_xor(acc, o, 32);
  if (lane == 0) out[b * 512 + n] = acc + outb[0];
}

// ---------------- host launcher ----------------
extern "C" void kernel_launch(void* const* d_in, const int* in_sizes, int n_in,
                              void* d_out, int out_size, void* d_ws, size_t ws_size,
                              hipStream_t stream) {
  (void)in_sizes; (void)n_in; (void)out_size; (void)ws_size;
  const float* x      = (const float*)d_in[0];
  const float* prior  = (const float*)d_in[1];
  const float* projW  = (const float*)d_in[2];
  const float* projb  = (const float*)d_in[3];
  const float* Wih    = (const float*)d_in[4];
  const float* Whh    = (const float*)d_in[5];
  const float* bih    = (const float*)d_in[6];
  const float* bhh    = (const float*)d_in[7];
  const float* spW    = (const float*)d_in[8];
  const float* spB    = (const float*)d_in[9];
  const float* rW     = (const float*)d_in[10];
  const float* rb     = (const float*)d_in[11];
  const float* outW   = (const float*)d_in[12];
  const float* outb   = (const float*)d_in[13];
  const float* plog   = (const float*)d_in[14];
  const float* quer   = (const float*)d_in[15];
  float* out = (float*)d_out;

  size_t off = 0;
  auto alloc = [&](size_t bytes) -> size_t {
    size_t o = off;
    off = (off + bytes + 255) & ~(size_t)255;
    return o;
  };
  char* ws = (char*)d_ws;
  float*    w_u      = (float*)(ws + alloc(384 * 4));
  float*    w_v      = (float*)(ws + alloc(384 * 4));
  _Float16* w_whh16  = (_Float16*)(ws + alloc(49152 * 2));
  _Float16* w_pr16   = (_Float16*)(ws + alloc(262144 * 2));
  float*    w_qn     = (float*)(ws + alloc(12288 * 4));
  float*    w_h      = (float*)(ws + alloc((size_t)4096 * 128 * 4));
  _Float16* w_hT     = (_Float16*)(ws + alloc((size_t)4096 * 128 * 2));
  float*    w_zp     = (float*)(ws + alloc((size_t)4096 * 128 * 4));
  float*    w_scores = (float*)(ws + alloc((size_t)768 * 512 * 4));
  float*    w_probs  = (float*)(ws + alloc(24 * 4));
  int*      w_hidx   = (int*)(ws + alloc(768 * 25 * 4));
  float*    w_hw     = (float*)(ws + alloc(768 * 25 * 4));
  float*    w_hmix   = (float*)(ws + alloc((size_t)131072 * 4));
  float*    w_edge   = (float*)(ws + alloc((size_t)8 * 32 * 128 * 4));

  // setup
  uv_kernel<<<dim3(3), dim3(128), 0, stream>>>(Wih, projW, projb, bih, w_u, w_v);
  cvt_f16_kernel<<<dim3(192), dim3(256), 0, stream>>>(Whh, w_whh16, 49152);
  cvt_f16_kernel<<<dim3(1024), dim3(256), 0, stream>>>(prior, w_pr16, 262144);
  qnorm_kernel<<<dim3(12), dim3(256), 0, stream>>>(quer, w_qn);

  // GRU (WMMA)
  gru_kernel<<<dim3(32), dim3(256), 0, stream>>>(x, w_u, w_v, bhh, w_whh16, w_h, w_hT);

  // router
  stats_kernel<<<dim3(8), dim3(256), 0, stream>>>(w_h, rW, rb, w_probs, out + 4096);

  // z_prior (WMMA)
  zprior_kernel<<<dim3(512), dim3(128), 0, stream>>>(w_pr16, w_hT, w_zp);

  // scores / topk / mix / edge / out
  scores_kernel<<<dim3(1536), dim3(256), 0, stream>>>(w_h, spW, spB, w_qn, w_scores);
  topk_kernel<<<dim3(96), dim3(256), 0, stream>>>(w_scores, w_hidx, w_hw);
  hmix_kernel<<<dim3(512), dim3(256), 0, stream>>>(w_hidx, w_hw, w_probs, w_hmix);
  edge_kernel<<<dim3(256), dim3(128), 0, stream>>>(w_hmix, w_h, w_edge);
  final_kernel<<<dim3(512), dim3(256), 0, stream>>>(w_hmix, w_edge, w_zp, outW, outb, plog, out);
}
